// SentimentClassificationMoE_14611478741706
// MI455X (gfx1250) — compile-verified, matrix operand order checked by reading, not run on hardware
//
#include <hip/hip_runtime.h>
#include <math.h>

// ---------------- problem constants ----------------
#define BTOK   8192
#define SEQ    64
#define EMB    512
#define NEXP   16
#define FFN    2048
#define CAP    512     // BTOK / NEXP
#define NOUT   2

typedef __attribute__((ext_vector_type(16))) _Float16 v16h;
typedef __attribute__((ext_vector_type(8)))  _Float16 v8h;
typedef __attribute__((ext_vector_type(2)))  _Float16 v2h;
typedef __attribute__((ext_vector_type(8)))  float    v8f;

union V16H { v16h v; v8h h[2]; };

// ---------------- wave helpers (wave32) ----------------
__device__ inline float shfl_xor_f(float v, int m) {
    int lane = (int)(threadIdx.x & 31);
    return __int_as_float(__builtin_amdgcn_ds_bpermute(((lane ^ m) << 2), __float_as_int(v)));
}
__device__ inline int shfl_xor_i(int v, int m) {
    int lane = (int)(threadIdx.x & 31);
    return __builtin_amdgcn_ds_bpermute(((lane ^ m) << 2), v);
}
__device__ inline unsigned ballot32(bool p) {
    return __builtin_amdgcn_ballot_w32(p);
}

// ---------------- 1) embedding gather + mean pool ----------------
// xm[b,:] = mean_s emb[x[b,s],:]   (1.07 GB streamed: the HBM-bound stage)
__global__ __launch_bounds__(256) void embed_mean_kernel(
    const int* __restrict__ x, const float* __restrict__ emb, float* __restrict__ xm)
{
    int b = blockIdx.x;
    int t = threadIdx.x;                 // handles dims 2t, 2t+1
    const int* xr = x + (size_t)b * SEQ;
    float ax = 0.f, ay = 0.f;
    for (int s = 0; s < SEQ; ++s) {
        int row = xr[s];
        const float2* p = (const float2*)(emb + (size_t)row * EMB);
        float2 v = p[t];
        ax += v.x; ay += v.y;
    }
    const float inv = 1.0f / (float)SEQ;
    float2* o = (float2*)(xm + (size_t)b * EMB);
    o[t] = make_float2(ax * inv, ay * inv);
}

// ---------------- 2) weight f32 -> f16 transpose (B-fragment-friendly [N][K]) ----
// in: [E][R][C] f32 row-major; out: [E][C][R] f16 row-major
__global__ __launch_bounds__(256) void transpose_f32_to_f16(
    const float* __restrict__ in, _Float16* __restrict__ out, int R, int C)
{
    __shared__ float tile[32][33];
    int e = blockIdx.z;
    const float* inp = in + (size_t)e * R * C;
    _Float16*   outp = out + (size_t)e * R * C;
    int c0 = blockIdx.x * 32, r0 = blockIdx.y * 32;
    int tx = threadIdx.x, ty = threadIdx.y;      // blockDim (32,8)
    #pragma unroll
    for (int j = 0; j < 32; j += 8)
        tile[ty + j][tx] = inp[(size_t)(r0 + ty + j) * C + (c0 + tx)];
    __syncthreads();
    #pragma unroll
    for (int j = 0; j < 32; j += 8)
        outp[(size_t)(c0 + ty + j) * R + (r0 + tx)] = (_Float16)tile[tx][ty + j];
}

// ---------------- 3) gating: logits -> argmax + softmax prob of winner ------
// 16 threads per token (one per expert); butterfly reductions over 16 lanes
__global__ __launch_bounds__(256) void gate_kernel(
    const float* __restrict__ xm, const float* __restrict__ wg,
    int* __restrict__ eidx, float* __restrict__ gate_s)
{
    int tid = threadIdx.x;
    int grp = tid >> 4;          // 16 tokens / block
    int e   = tid & 15;
    int b = blockIdx.x * 16 + grp;
    const float* xr = xm + (size_t)b * EMB;
    float logit = 0.f;
    for (int k = 0; k < EMB; ++k)
        logit = fmaf(xr[k], wg[(size_t)k * NEXP + e], logit);
    // argmax (first-max tie break) over the 16-lane group
    float mv = logit; int mi = e;
    #pragma unroll
    for (int m = 1; m <= 8; m <<= 1) {
        float ov = shfl_xor_f(mv, m);
        int   oi = shfl_xor_i(mi, m);
        if (ov > mv || (ov == mv && oi < mi)) { mv = ov; mi = oi; }
    }
    // softmax denom;  winner prob = exp(0)/sum = 1/sum
    float ex = expf(logit - mv);
    #pragma unroll
    for (int m = 1; m <= 8; m <<= 1) ex += shfl_xor_f(ex, m);
    if (e == 0) { eidx[b] = mi; gate_s[b] = 1.0f / ex; }
}

// ---------------- 4) init helper ----------------
__global__ void init_i32(int* p, int v, int n) {
    int i = blockIdx.x * blockDim.x + threadIdx.x;
    if (i < n) p[i] = v;
}

// ---------------- 5) deterministic capacity scan (batch-ordered cumsum) ------
// slot(b) = #earlier tokens routed to same expert; drop if slot >= CAP
__global__ __launch_bounds__(256) void scan_kernel(
    const int* __restrict__ eidx, int* __restrict__ pos, int* __restrict__ token_of)
{
    __shared__ int base[NEXP];
    __shared__ int wcnt[8][NEXP];
    __shared__ int woff[8][NEXP];
    int tid = threadIdx.x, lane = tid & 31, wave = tid >> 5;
    if (tid < NEXP) base[tid] = 0;
    __syncthreads();
    unsigned lower = (lane == 0) ? 0u : ((1u << lane) - 1u);
    for (int chunk = 0; chunk < BTOK; chunk += 256) {
        int b = chunk + tid;
        int e = eidx[b];
        int myrank = 0;
        for (int ee = 0; ee < NEXP; ++ee) {
            unsigned bal = ballot32(e == ee);
            if (ee == e) myrank = __popc(bal & lower);
            if (lane == 0) wcnt[wave][ee] = __popc(bal);
        }
        __syncthreads();
        if (tid < NEXP) {
            int run = base[tid];
            #pragma unroll
            for (int w = 0; w < 8; ++w) { int c = wcnt[w][tid]; woff[w][tid] = run; run += c; }
            base[tid] = run;
        }
        __syncthreads();
        int slot = woff[wave][e] + myrank;
        if (slot < CAP) { pos[b] = slot; token_of[e * CAP + slot] = b; }
        else            { pos[b] = -1; }
        __syncthreads();
    }
}

// ---------------- 6) dispatch: gather kept tokens into f16 A-matrix ---------
__global__ __launch_bounds__(256) void dispatch_kernel(
    const float* __restrict__ xm, const int* __restrict__ token_of,
    _Float16* __restrict__ xeh)
{
    int slot = blockIdx.x;          // e*CAP + c
    int tid  = threadIdx.x;         // handles dims 2t, 2t+1
    int t = token_of[slot];
    v2h o;
    if (t >= 0) {
        const float2* p = (const float2*)(xm + (size_t)t * EMB);
        float2 v = p[tid];
        o[0] = (_Float16)v.x; o[1] = (_Float16)v.y;
    } else {
        o[0] = (_Float16)0.f; o[1] = (_Float16)0.f;
    }
    *(v2h*)(xeh + (size_t)slot * EMB + (size_t)tid * 2) = o;
}

// ---------------- 7) WMMA GEMM: C[e] = act(A[e] @ Bt[e]^T + bias[e]) --------
// A: [E][M][K] f16 row-major. Bt: [E][N][K] f16 (pre-transposed weights).
// Fragment layout per ISA 16-bit A/B: lanes 0-15 hold K{0..7,16..23},
// lanes 16-31 hold K{8..15,24..31} -> two contiguous 16B loads per lane.
__device__ inline v16h ldfrag(const _Float16* p) {
    V16H u;
    u.h[0] = *(const v8h*)(p);
    u.h[1] = *(const v8h*)(p + 16);
    return u.v;
}

template<bool RELU_F16OUT>
__global__ __launch_bounds__(256) void wmma_gemm_kernel(
    const _Float16* __restrict__ A, const _Float16* __restrict__ Bt,
    const float* __restrict__ bias, float* __restrict__ Cf, _Float16* __restrict__ Ch,
    int M, int N, int K)
{
    const int e    = blockIdx.z;
    const int tid  = threadIdx.x;
    const int wave = tid >> 5, lane = tid & 31;
    const int l    = lane & 15, hi = lane >> 4;
    // block tile 128x128; 8 waves in 4(m) x 2(n); wave tile 32x64
    const int mWave = blockIdx.y * 128 + (wave >> 1) * 32;
    const int nWave = blockIdx.x * 128 + (wave & 1) * 64;

    const _Float16* Ae = A  + (size_t)e * M * K;
    const _Float16* Be = Bt + (size_t)e * N * K;

    const _Float16* aP0 = Ae + (size_t)(mWave + l) * K + hi * 8;
    const _Float16* aP1 = aP0 + (size_t)16 * K;
    const _Float16* bP0 = Be + (size_t)(nWave + l) * K + hi * 8;
    const _Float16* bP1 = bP0 + (size_t)16 * K;
    const _Float16* bP2 = bP0 + (size_t)32 * K;
    const _Float16* bP3 = bP0 + (size_t)48 * K;

    v8f acc[2][4];
    #pragma unroll
    for (int i = 0; i < 2; ++i)
        #pragma unroll
        for (int j = 0; j < 4; ++j) acc[i][j] = {};

    for (int k0 = 0; k0 < K; k0 += 32) {
        v16h a0 = ldfrag(aP0 + k0);
        v16h a1 = ldfrag(aP1 + k0);
        v16h b0 = ldfrag(bP0 + k0);
        v16h b1 = ldfrag(bP1 + k0);
        v16h b2 = ldfrag(bP2 + k0);
        v16h b3 = ldfrag(bP3 + k0);
        acc[0][0] = __builtin_amdgcn_wmma_f32_16x16x32_f16(false, a0, false, b0, (short)0, acc[0][0], false, false);
        acc[0][1] = __builtin_amdgcn_wmma_f32_16x16x32_f16(false, a0, false, b1, (short)0, acc[0][1], false, false);
        acc[0][2] = __builtin_amdgcn_wmma_f32_16x16x32_f16(false, a0, false, b2, (short)0, acc[0][2], false, false);
        acc[0][3] = __builtin_amdgcn_wmma_f32_16x16x32_f16(false, a0, false, b3, (short)0, acc[0][3], false, false);
        acc[1][0] = __builtin_amdgcn_wmma_f32_16x16x32_f16(false, a1, false, b0, (short)0, acc[1][0], false, false);
        acc[1][1] = __builtin_amdgcn_wmma_f32_16x16x32_f16(false, a1, false, b1, (short)0, acc[1][1], false, false);
        acc[1][2] = __builtin_amdgcn_wmma_f32_16x16x32_f16(false, a1, false, b2, (short)0, acc[1][2], false, false);
        acc[1][3] = __builtin_amdgcn_wmma_f32_16x16x32_f16(false, a1, false, b3, (short)0, acc[1][3], false, false);
    }

    // epilogue: C layout VGPR r -> M = r + (lane<16 ? 0 : 8), N = l
    #pragma unroll
    for (int mi = 0; mi < 2; ++mi) {
        int mrow = mWave + mi * 16 + hi * 8;
        #pragma unroll
        for (int j = 0; j < 4; ++j) {
            int n = nWave + j * 16 + l;
            float bv = bias[(size_t)e * N + n];
            #pragma unroll
            for (int r = 0; r < 8; ++r) {
                float v = acc[mi][j][r] + bv;
                if constexpr (RELU_F16OUT) {
                    Ch[((size_t)e * M + mrow + r) * N + n] = (_Float16)fmaxf(v, 0.0f);
                } else {
                    Cf[((size_t)e * M + mrow + r) * N + n] = v;
                }
            }
        }
    }
}

// ---------------- 8) gate-weighted combine + fc + log_softmax ----------------
__global__ __launch_bounds__(256) void combine_fc_kernel(
    const float* __restrict__ ye, const float* __restrict__ gate_s,
    const int* __restrict__ eidx, const int* __restrict__ pos,
    const float* __restrict__ fcw, const float* __restrict__ fcb,
    float* __restrict__ out)
{
    int tid = threadIdx.x, lane = tid & 31, grp = tid >> 5;  // 8 tokens / block
    int b = blockIdx.x * 8 + grp;
    float a0 = 0.f, a1 = 0.f;
    int p = pos[b];
    if (p >= 0) {
        int e = eidx[b];
        float g = gate_s[b];
        const float* yr = ye + ((size_t)e * CAP + p) * EMB;
        #pragma unroll
        for (int i = 0; i < EMB / 32; ++i) {
            int d = lane + i * 32;
            float v = g * yr[d];
            a0 = fmaf(v, fcw[(size_t)d * NOUT + 0], a0);
            a1 = fmaf(v, fcw[(size_t)d * NOUT + 1], a1);
        }
    }
    #pragma unroll
    for (int m = 16; m >= 1; m >>= 1) { a0 += shfl_xor_f(a0, m); a1 += shfl_xor_f(a1, m); }
    if (lane == 0) {
        float z0 = a0 + fcb[0], z1 = a1 + fcb[1];
        float mx = fmaxf(z0, z1);
        float lse = mx + logf(expf(z0 - mx) + expf(z1 - mx));
        out[(size_t)b * NOUT + 0] = z0 - lse;
        out[(size_t)b * NOUT + 1] = z1 - lse;
    }
}

// ---------------- launch ----------------
extern "C" void kernel_launch(void* const* d_in, const int* in_sizes, int n_in,
                              void* d_out, int out_size, void* d_ws, size_t ws_size,
                              hipStream_t stream) {
    const int*   x   = (const int*)d_in[0];
    const float* emb = (const float*)d_in[1];
    const float* wg  = (const float*)d_in[2];
    const float* w1  = (const float*)d_in[3];
    const float* b1  = (const float*)d_in[4];
    const float* w2  = (const float*)d_in[5];
    const float* b2  = (const float*)d_in[6];
    const float* fcw = (const float*)d_in[7];
    const float* fcb = (const float*)d_in[8];
    float* out = (float*)d_out;

    char* ws = (char*)d_ws;
    size_t off = 0;
    auto wsalloc = [&](size_t bytes) -> void* {
        void* p = ws + off;
        off = (off + bytes + 255) & ~(size_t)255;
        return p;
    };
    float*     xm    = (float*)    wsalloc((size_t)BTOK * EMB * 4);           // 16 MB
    _Float16*  w1t   = (_Float16*) wsalloc((size_t)NEXP * FFN * EMB * 2);     // 32 MB  [E][FFN][EMB]
    _Float16*  w2t   = (_Float16*) wsalloc((size_t)NEXP * EMB * FFN * 2);     // 32 MB  [E][EMB][FFN]
    _Float16*  xeh   = (_Float16*) wsalloc((size_t)NEXP * CAP * EMB * 2);     // 8 MB   [E][CAP][EMB]
    _Float16*  hbuf  = (_Float16*) wsalloc((size_t)NEXP * CAP * FFN * 2);     // 32 MB  [E][CAP][FFN]
    float*     ye    = (float*)    wsalloc((size_t)NEXP * CAP * EMB * 4);     // 16 MB
    float*     gate  = (float*)    wsalloc((size_t)BTOK * 4);
    int*       eidx  = (int*)      wsalloc((size_t)BTOK * 4);
    int*       pos   = (int*)      wsalloc((size_t)BTOK * 4);
    int*       tknof = (int*)      wsalloc((size_t)NEXP * CAP * 4);

    // 1) embedding + mean pool (HBM-bound, ~1.07 GB)
    embed_mean_kernel<<<BTOK, 256, 0, stream>>>(x, emb, xm);
    // 2) one-time weight convert+transpose f32 -> f16 [N][K]
    transpose_f32_to_f16<<<dim3(FFN / 32, EMB / 32, NEXP), dim3(32, 8), 0, stream>>>(w1, w1t, EMB, FFN);
    transpose_f32_to_f16<<<dim3(EMB / 32, FFN / 32, NEXP), dim3(32, 8), 0, stream>>>(w2, w2t, FFN, EMB);
    // 3) gating
    gate_kernel<<<BTOK / 16, 256, 0, stream>>>(xm, wg, eidx, gate);
    // 4-5) deterministic capacity assignment
    init_i32<<<(NEXP * CAP + 255) / 256, 256, 0, stream>>>(tknof, -1, NEXP * CAP);
    scan_kernel<<<1, 256, 0, stream>>>(eidx, pos, tknof);
    // 6) dispatch to f16 A-matrix
    dispatch_kernel<<<NEXP * CAP, 256, 0, stream>>>(xm, tknof, xeh);
    // 7) expert FFN via WMMA: relu(Xe@W1+b1) then @W2+b2
    wmma_gemm_kernel<true><<<dim3(FFN / 128, CAP / 128, NEXP), 256, 0, stream>>>(
        xeh, w1t, b1, nullptr, hbuf, CAP, FFN, EMB);
    wmma_gemm_kernel<false><<<dim3(EMB / 128, CAP / 128, NEXP), 256, 0, stream>>>(
        hbuf, w2t, b2, ye, nullptr, CAP, EMB, FFN);
    // 8) combine + classifier + log_softmax
    combine_fc_kernel<<<BTOK / 8, 256, 0, stream>>>(ye, gate, eidx, pos, fcw, fcb, out);
}
